// Crossatten_align_atttrans_53893249630470
// MI455X (gfx1250) — matrix operand order repaired and checked by reading, not compiled, not run on hardware
//
#include <hip/hip_runtime.h>
#include <hip/hip_bf16.h>

// ---------------------------------------------------------------------------
// Fused window cross-attention for MI455X (gfx1250, wave32, WMMA).
// One workgroup (8 wave32) per 7x7 window; everything staged in LDS; math in
// v_wmma_f32_16x16x32_bf16. Only HBM traffic: read x/y, write out + attn.
// ---------------------------------------------------------------------------

typedef __attribute__((ext_vector_type(16))) __bf16 v16bf;
typedef __attribute__((ext_vector_type(8)))  __bf16 v8bf;
typedef __attribute__((ext_vector_type(8)))  float  v8f;

#define WS 7
#define NN 49          // tokens per window (padded to 64)
#define C  256
#define NH 8
#define HD 32
#define QSCALE 0.17677669529663687f   // 32^-0.5

__device__ __forceinline__ __bf16 to_bf16(float f) {
    unsigned u = __builtin_bit_cast(unsigned, f);
    unsigned r = u + 0x7fffu + ((u >> 16) & 1u);   // round-to-nearest-even
    return __builtin_bit_cast(__bf16, (unsigned short)(r >> 16));
}

__device__ __forceinline__ v8f zero8() {
    v8f z = {0.f, 0.f, 0.f, 0.f, 0.f, 0.f, 0.f, 0.f};
    return z;
}

// A-fragment (16x32 bf16, M x K): lane group g=lane/16, row m=lane%16.
// halves 0..7  <- K = kbase + g*8 + i      (contiguous 16B)
// halves 8..15 <- K = kbase + 16 + g*8 + i (contiguous 16B)
__device__ __forceinline__ v16bf ldA(const __bf16* row, int kb) {
    v8bf lo = *(const v8bf*)(row + kb);
    v8bf hi = *(const v8bf*)(row + kb + 16);
    v16bf r;
#pragma unroll
    for (int i = 0; i < 8; ++i) { r[i] = lo[i]; r[i + 8] = hi[i]; }
    return r;
}

// B-fragment (32x16 bf16, K x N): lane col n=lane%16, group g=lane/16.
// halves 0..7 <- K = kbase + g*8 + i, halves 8..15 <- K = kbase + 16 + g*8 + i.  p already points at
// the start of the 16-element run for this lane.
__device__ __forceinline__ v16bf ldB(const __bf16* p) {
    v8bf lo = *(const v8bf*)(p);
    v8bf hi = *(const v8bf*)(p + 16);
    v16bf r;
#pragma unroll
    for (int i = 0; i < 8; ++i) { r[i] = lo[i]; r[i + 8] = hi[i]; }
    return r;
}

__device__ __forceinline__ v8f wmma_bf16(v16bf a, v16bf b, v8f c) {
    return __builtin_amdgcn_wmma_f32_16x16x32_bf16(
        /*neg_a=*/false, a, /*neg_b=*/false, b,
        /*c_mod=*/(short)0, c, /*reuse_a=*/false, /*reuse_b=*/false);
}

// 64x256 = A(64x256 LDS bf16) * W^T (W is [256 out][256 in] bf16, row-major).
// 64 output tiles; each of 8 waves owns 8 tiles: mt in 0..3, nt = wave*2+{0,1}.
__device__ __forceinline__ void gemm64x256(const __bf16* __restrict__ A,
                                           const __bf16* __restrict__ W,
                                           v8f acc[8], int tid) {
    const int wv = tid >> 5, l = tid & 31, g = l >> 4, lm = l & 15;
#pragma unroll
    for (int kk = 0; kk < 8; ++kk) {
        const int kb = kk * 32;
        v16bf a[4], b[2];
#pragma unroll
        for (int mt = 0; mt < 4; ++mt)
            a[mt] = ldA(A + (mt * 16 + lm) * C, kb + g * 8);
#pragma unroll
        for (int j = 0; j < 2; ++j) {
            int n = (wv * 2 + j) * 16 + lm;
            b[j] = ldB(W + n * C + kb + g * 8);
        }
#pragma unroll
        for (int mt = 0; mt < 4; ++mt)
#pragma unroll
            for (int j = 0; j < 2; ++j)
                acc[mt * 2 + j] = wmma_bf16(a[mt], b[j], acc[mt * 2 + j]);
    }
}

// Store the 8 per-wave 16x16 f32 D tiles as bf16 (optionally transposed into a
// [256][64] buffer).  C/D layout: vgpr r -> row = mt*16 + (lane/16)*8 + r,
// col = nt*16 + lane%16.
__device__ __forceinline__ void store64x256(const v8f acc[8],
                                            const float* __restrict__ bias,
                                            float scale, __bf16* dst,
                                            bool transpose, int tid) {
    const int wv = tid >> 5, l = tid & 31, g = l >> 4, lm = l & 15;
#pragma unroll
    for (int i = 0; i < 8; ++i) {
        int mt = i >> 1, nt = wv * 2 + (i & 1);
        int col = nt * 16 + lm;
        float bz = bias[col];
#pragma unroll
        for (int r = 0; r < 8; ++r) {
            int row = mt * 16 + g * 8 + r;
            float v = (acc[i][r] + bz) * scale;
            if (transpose) dst[col * 64 + row] = to_bf16(v);
            else           dst[row * C  + col] = to_bf16(v);
        }
    }
}

// Convert the four 256x256 fp32 weight matrices to bf16 in workspace (L2-hot).
__global__ __launch_bounds__(256) void convert_weights_kernel(
    const float* __restrict__ Wq, const float* __restrict__ Wk,
    const float* __restrict__ Wv, const float* __restrict__ Wp,
    __bf16* __restrict__ dst) {
    int i = blockIdx.x * 256 + threadIdx.x;            // 65536 elems / matrix
    dst[i]          = to_bf16(Wq[i]);
    dst[i +  65536] = to_bf16(Wk[i]);
    dst[i + 131072] = to_bf16(Wv[i]);
    dst[i + 196608] = to_bf16(Wp[i]);
}

__global__ __launch_bounds__(256) void fused_window_attn_kernel(
    const float* __restrict__ x, const float* __restrict__ y,
    const float* __restrict__ bq, const float* __restrict__ bk,
    const float* __restrict__ bv, const float* __restrict__ bp,
    const float* __restrict__ bias_table, const __bf16* __restrict__ wbf,
    float* __restrict__ out, float* __restrict__ attn_out) {
    // LDS (~150 KB total; gfx1250 CU-mode allows 160 KB):
    __shared__ __align__(32) __bf16 s_xq[64 * C];   // masked-x, then Q
    __shared__ __align__(32) __bf16 s_yv[64 * C];   // y, then V^T ([256][64])
    __shared__ __align__(32) __bf16 s_k [64 * C];   // K
    __shared__ __align__(32) __bf16 s_o [64 * C];   // attention output O
    __shared__ float s_S[64 * 64];                  // per-head scores / probs
    __shared__ float s_mw[64];                      // exposure mask per token
    __shared__ float s_bias[169 * NH];              // relative-position bias

    const int tid = threadIdx.x;
    const int wv = tid >> 5, l = tid & 31, g = l >> 4, lm = l & 15;
    const int win = blockIdx.x;                 // 0..4095
    const int bb  = win >> 10;                  // batch
    const int wh  = (win >> 5) & 31;            // window row
    const int ww  = win & 31;                   // window col

    // ---- stage relative-position bias table into LDS ----
    for (int i = tid; i < 169 * NH; i += 256) s_bias[i] = bias_table[i];

    // ---- stage 1: load x/y window, compute exposure mask, pack bf16 ----
#pragma unroll
    for (int rr = 0; rr < 8; ++rr) {
        int tok = wv * 8 + rr;                  // 0..63
        if (tok < NN) {
            int gy = wh * WS + tok / WS, gx = ww * WS + tok % WS;
            size_t base = (((size_t)bb * 224 + gy) * 224 + gx) * C;
            const float4* px = (const float4*)(x + base + l * 8);
            const float4* py = (const float4*)(y + base + l * 8);
            float4 x0 = px[0], x1 = px[1];
            float4 y0 = py[0], y1 = py[1];
            int cnt = (x0.x > 0.95f) + (x0.y > 0.95f) + (x0.z > 0.95f) + (x0.w > 0.95f)
                    + (x1.x > 0.95f) + (x1.y > 0.95f) + (x1.z > 0.95f) + (x1.w > 0.95f);
#pragma unroll
            for (int off = 16; off; off >>= 1) cnt += __shfl_xor(cnt, off, 32);
            float m = 1.0f - (float)cnt * (1.0f / 256.0f);
            if (l == 0) s_mw[tok] = m;
            __bf16* xr = s_xq + tok * C + l * 8;
            __bf16* yr = s_yv + tok * C + l * 8;
            xr[0] = to_bf16(x0.x * m); xr[1] = to_bf16(x0.y * m);
            xr[2] = to_bf16(x0.z * m); xr[3] = to_bf16(x0.w * m);
            xr[4] = to_bf16(x1.x * m); xr[5] = to_bf16(x1.y * m);
            xr[6] = to_bf16(x1.z * m); xr[7] = to_bf16(x1.w * m);
            yr[0] = to_bf16(y0.x); yr[1] = to_bf16(y0.y);
            yr[2] = to_bf16(y0.z); yr[3] = to_bf16(y0.w);
            yr[4] = to_bf16(y1.x); yr[5] = to_bf16(y1.y);
            yr[6] = to_bf16(y1.z); yr[7] = to_bf16(y1.w);
        } else {
            if (l == 0) s_mw[tok] = 0.f;
            __bf16 z = to_bf16(0.f);
#pragma unroll
            for (int e = 0; e < 8; ++e) {
                s_xq[tok * C + l * 8 + e] = z;
                s_yv[tok * C + l * 8 + e] = z;
            }
        }
    }
    __syncthreads();

    // ---- stage 2: K = y @ Wk^T + bk (into its own buffer) ----
    {
        v8f kacc[8];
#pragma unroll
        for (int i = 0; i < 8; ++i) kacc[i] = zero8();
        gemm64x256(s_yv, wbf + 65536, kacc, tid);
        store64x256(kacc, bk, 1.0f, s_k, false, tid);
    }

    // ---- stage 3: Q and V accumulated in regs, then written over x/y LDS ----
    {
        v8f qacc[8], vacc[8];
#pragma unroll
        for (int i = 0; i < 8; ++i) { qacc[i] = zero8(); vacc[i] = zero8(); }
        gemm64x256(s_xq, wbf,               qacc, tid);   // Q reads masked x
        gemm64x256(s_yv, wbf + 2 * 65536,   vacc, tid);   // V reads y
        __syncthreads();                                  // all reads done
        store64x256(qacc, bq, QSCALE, s_xq, false, tid);  // Q  -> [64][256]
        store64x256(vacc, bv, 1.0f,   s_yv, true,  tid);  // V^T-> [256][64]
    }

    // ---- stage 4: per-head attention ----
    for (int h = 0; h < NH; ++h) {
        __syncthreads();   // Q/K/V^T visible; previous head's P fully consumed

        // S = Q_h K_h^T : 16 tiles of 16x16, one 16x16x32 WMMA each (K=hd=32)
#pragma unroll
        for (int j = 0; j < 2; ++j) {
            int t = wv * 2 + j;
            int mt = t >> 2, nt = t & 3;
            v16bf a  = ldA(s_xq + (mt * 16 + lm) * C, h * HD + g * 8);
            v16bf bm = ldB(s_k  + (nt * 16 + lm) * C + h * HD + g * 8);
            v8f z = zero8();
            z = wmma_bf16(a, bm, z);
#pragma unroll
            for (int r = 0; r < 8; ++r)
                s_S[(mt * 16 + g * 8 + r) * 64 + nt * 16 + lm] = z[r];
        }
        __syncthreads();

        // mask * bias * softmax over the 49 valid keys; also emit attn output
        if (tid < 64) {
            int a = tid;
            float* srow = s_S + a * 64;
            if (a < NN) {
                float mwa = s_mw[a];
                int ia = a / WS, ja = a % WS;
                float mx = -1e30f;
                for (int b2 = 0; b2 < NN; ++b2) {
                    int idx = (ia - b2 / WS + WS - 1) * (2 * WS - 1)
                            + (ja - b2 % WS + WS - 1);
                    float s = srow[b2] * mwa * s_mw[b2] + s_bias[idx * NH + h];
                    srow[b2] = s;
                    mx = fmaxf(mx, s);
                }
                float sum = 0.f;
                for (int b2 = 0; b2 < NN; ++b2) {
                    float e = __expf(srow[b2] - mx);
                    srow[b2] = e; sum += e;
                }
                float inv = 1.f / sum;
                float* arow = attn_out + (((size_t)win * NH + h) * NN + a) * NN;
                for (int b2 = 0; b2 < NN; ++b2) {
                    float p = srow[b2] * inv;
                    srow[b2] = p;
                    arow[b2] = p;
                }
                for (int b2 = NN; b2 < 64; ++b2) srow[b2] = 0.f;
            } else {
                for (int b2 = 0; b2 < 64; ++b2) srow[b2] = 0.f;
            }
        }
        __syncthreads();

        // O_h = P @ V_h : 8 tiles (4 M x 2 N16), K = 64 -> two 32-K steps
        {
            int mt = wv & 3, nch = wv >> 2;
            v8f oacc = zero8();
#pragma unroll
            for (int kk = 0; kk < 2; ++kk) {
                const float* prow = s_S + (mt * 16 + lm) * 64 + kk * 32 + g * 8;
                v16bf a;
#pragma unroll
                for (int i2 = 0; i2 < 8; ++i2) {
                    a[i2]     = to_bf16(prow[i2]);
                    a[i2 + 8] = to_bf16(prow[16 + i2]);
                }
                int ch = h * HD + nch * 16 + lm;          // V^T row = channel
                v16bf bm = ldB(s_yv + ch * 64 + kk * 32 + g * 8);
                oacc = wmma_bf16(a, bm, oacc);
            }
#pragma unroll
            for (int r = 0; r < 8; ++r)
                s_o[(mt * 16 + g * 8 + r) * C + h * HD + nch * 16 + lm] =
                    to_bf16(oacc[r]);
        }
    }
    __syncthreads();

    // ---- stage 5: out = O @ Wp^T + bp, window-reverse scatter to HBM ----
    {
        v8f pacc[8];
#pragma unroll
        for (int i = 0; i < 8; ++i) pacc[i] = zero8();
        gemm64x256(s_o, wbf + 3 * 65536, pacc, tid);
#pragma unroll
        for (int i = 0; i < 8; ++i) {
            int mt = i >> 1, nt = wv * 2 + (i & 1);
            int col = nt * 16 + lm;
            float bz = bp[col];
#pragma unroll
            for (int r = 0; r < 8; ++r) {
                int tok = mt * 16 + g * 8 + r;
                if (tok < NN) {
                    int gy = wh * WS + tok / WS, gx = ww * WS + tok % WS;
                    size_t o = (((size_t)bb * 224 + gy) * 224 + gx) * C + col;
                    out[o] = pacc[i][r] + bz;
                }
            }
        }
    }
}

extern "C" void kernel_launch(void* const* d_in, const int* in_sizes, int n_in,
                              void* d_out, int out_size, void* d_ws, size_t ws_size,
                              hipStream_t stream) {
    const float* x          = (const float*)d_in[0];
    const float* y          = (const float*)d_in[1];
    const float* Wq         = (const float*)d_in[2];
    const float* bq         = (const float*)d_in[3];
    const float* Wk         = (const float*)d_in[4];
    const float* bk         = (const float*)d_in[5];
    const float* Wv         = (const float*)d_in[6];
    const float* bv         = (const float*)d_in[7];
    const float* Wp         = (const float*)d_in[8];
    const float* bp         = (const float*)d_in[9];
    const float* bias_table = (const float*)d_in[10];

    float* out      = (float*)d_out;
    float* attn_out = out + (size_t)4 * 224 * 224 * 256;   // second tuple output

    __bf16* wbf = (__bf16*)d_ws;                           // 512 KB bf16 weights

    convert_weights_kernel<<<256, 256, 0, stream>>>(Wq, Wk, Wv, Wp, wbf);
    fused_window_attn_kernel<<<4096, 256, 0, stream>>>(
        x, y, bq, bk, bv, bp, bias_table, wbf, out, attn_out);
}
